// FeatureAdaptionCls_59803124629818
// MI455X (gfx1250) — matrix operand order repaired and checked by reading, not compiled, not run on hardware
//
#include <hip/hip_runtime.h>
#include <hip/hip_bf16.h>

typedef __attribute__((ext_vector_type(16))) __bf16 v16bf;
typedef __attribute__((ext_vector_type(8)))  float  v8f;

#define KT_TOTAL 72   // K = 2304 / 32 chunks

static __device__ __forceinline__ __bf16 f2bf(float f) { return (__bf16)f; }

// ---------------------------------------------------------------------------
// Pack f32 OIHW weights into per-wave WMMA A-fragments (16x32 bf16).
// A-fragment k order used here (contiguous, matches B): lane holds M = lane&15;
//   lane<16 : K=e      (0..15)
//   lane>=16: K=e+16   (16..31)
// mode 0 (conv):   kt -> kk = kt>>3,  ch = (kt&7)*32 + K      (K-order: kk,c)
// mode 1 (deform): kt -> g=kt/18, r=kt%18, kk=r>>1, ch=g*64+(r&1)*32+K
// Fragment storage: dst[((mt*72+kt)*32 + lane)*16 + e]
// ---------------------------------------------------------------------------
__global__ void pack_wA(const float* __restrict__ src, __bf16* __restrict__ dst,
                        int outC, int MT, int mode)
{
    int idx = blockIdx.x * blockDim.x + threadIdx.x;
    int total = MT * KT_TOTAL * 512;
    if (idx >= total) return;
    int e    = idx & 15;
    int lane = (idx >> 4) & 31;
    int kt   = (idx >> 9) % KT_TOTAL;
    int mt   = idx / (KT_TOTAL * 512);
    int hi   = lane >> 4;
    int K    = e + 16 * hi;
    int m    = mt * 16 + (lane & 15);
    int kk, ch;
    if (mode == 0) { kk = kt >> 3; ch = ((kt & 7) << 5) + K; }
    else { int g = kt / 18, r = kt - g * 18; kk = r >> 1; ch = g * 64 + ((r & 1) << 5) + K; }
    float v = 0.0f;
    if (m < outC) v = src[(size_t)m * 2304 + (size_t)ch * 9 + kk];
    dst[idx] = f2bf(v);
}

// ---------------------------------------------------------------------------
// 3x3 SAME conv, 256 input channels, implicit GEMM on WMMA bf16.
// Block = 128 threads (4 waves). Tile: 64 outC (4x16) x 16 pixels (one row seg).
// grid.x = B*H*(W/16) = 2048 pixel tiles; grid.y = ceil(outCpad/64).
// ---------------------------------------------------------------------------
__global__ void __launch_bounds__(128)
conv3x3_wmma(const float* __restrict__ in, const __bf16* __restrict__ wpack,
             const float* __restrict__ bias, float* __restrict__ out, int outC)
{
    __shared__ __align__(32) __bf16 sIn[3 * 18 * 256];   // 27.6 KB halo tile, bf16

    int pt  = blockIdx.x;
    int w0  = (pt & 3) << 4;
    int h   = (pt >> 2) & 63;
    int b   = pt >> 8;
    int tid = threadIdx.x;
    int lane = tid & 31;
    int wv   = tid >> 5;

    // cooperative halo fill: rows h-1..h+1, cols w0-1..w0+16, all 256 channels
    for (int i = tid; i < 3 * 18 * 256; i += 128) {
        int wp = i % 18;
        int t  = i / 18;
        int yl = t % 3;
        int c  = t / 3;
        int gy = h + yl - 1;
        int gx = w0 + wp - 1;
        float fv = 0.0f;
        if (gy >= 0 && gy < 64 && gx >= 0 && gx < 64)
            fv = in[(((size_t)b * 256 + c) * 64 + gy) * 64 + gx];
        sIn[(yl * 18 + wp) * 256 + c] = f2bf(fv);
    }
    __syncthreads();

    int mt = blockIdx.y * 4 + wv;
    int p  = lane & 15;      // output pixel column within tile (B-frag N)
    int hi = lane >> 4;      // B-frag: lane<16 holds K0..15, lane>=16 holds K16..31

    v8f acc = {};
    const v16bf* apack = (const v16bf*)wpack;
    for (int kt = 0; kt < KT_TOTAL; ++kt) {
        int kk = kt >> 3;
        int ky = kk / 3, kx = kk - ky * 3;
        int c0 = (kt & 7) << 5;
        v16bf a  = apack[((size_t)mt * KT_TOTAL + kt) * 32 + lane];
        v16bf bm = *(const v16bf*)(&sIn[(ky * 18 + p + kx) * 256 + c0 + hi * 16]);
        acc = __builtin_amdgcn_wmma_f32_16x16x32_bf16(false, a, false, bm,
                                                      (short)0, acc, false, false);
    }

    // C/D layout: lane gives N = lane&15; row M = v + 8*(lane>>4)
    int n = w0 + p;
    for (int v = 0; v < 8; ++v) {
        int oc = mt * 16 + 8 * hi + v;
        if (oc < outC) {
            float r = acc[v] + bias[oc];
            out[(((size_t)b * outC + oc) * 64 + h) * 64 + n] = r;
        }
    }
}

// ---------------------------------------------------------------------------
// Deformable conv (dg=4, k=3) + ReLU as implicit GEMM on WMMA bf16.
// B fragments are built cooperatively in LDS: each of the 128 threads does one
// bilinear setup (its pixel/chunk) and samples 4 channels x 4 taps from global
// x (33.5 MB -> resident in 192 MB L2). Validity masks folded into tap weights
// per the reference semantics. 4 waves reuse each fragment for 4 M-tiles.
// ---------------------------------------------------------------------------
__global__ void __launch_bounds__(128)
deform_wmma(const float* __restrict__ x, const float* __restrict__ offs,
            const __bf16* __restrict__ wpack, float* __restrict__ out)
{
    __shared__ __align__(32) __bf16 sB[32 * 16];

    int pt  = blockIdx.x;
    int w0  = (pt & 3) << 4;
    int h   = (pt >> 2) & 63;
    int b   = pt >> 8;
    int tid = threadIdx.x;
    int lane = tid & 31;
    int wv   = tid >> 5;
    int mt   = blockIdx.y * 4 + wv;
    int p    = lane & 15;
    int hi   = lane >> 4;

    // this thread's role in B-fragment assembly: fragment lane blane, elems e0..e0+3
    int blane = tid >> 2;            // 0..31
    int e0    = (tid & 3) << 2;      // 0,4,8,12
    int bp    = blane & 15;
    int bhi   = blane >> 4;
    int wpix  = w0 + bp;

    v8f acc = {};
    const v16bf* apack = (const v16bf*)wpack;

    for (int kt = 0; kt < KT_TOTAL; ++kt) {
        int g  = kt / 18;
        int r  = kt - g * 18;
        int kk = r >> 1;
        int c0 = (r & 1) << 5;
        int ky = kk / 3, kx = kk - ky * 3;

        // bilinear setup for pixel (h, wpix), group g, tap kk
        float dy = offs[(((size_t)b * 72 + g * 18 + kk * 2    ) * 64 + h) * 64 + wpix];
        float dx = offs[(((size_t)b * 72 + g * 18 + kk * 2 + 1) * 64 + h) * 64 + wpix];
        float py = (float)(h + ky - 1) + dy;
        float px = (float)(wpix + kx - 1) + dx;
        float y0f = floorf(py), x0f = floorf(px);
        float wy1 = py - y0f, wy0 = 1.0f - wy1;
        float wx1 = px - x0f, wx0 = 1.0f - wx1;
        int y0 = (int)y0f, x0i = (int)x0f;
        int y1 = y0 + 1, x1i = x0i + 1;
        bool vy0 = (y0 >= 0) && (y0 < 64);
        bool vy1 = (y1 >= 0) && (y1 < 64);
        bool vx0 = (x0i >= 0) && (x0i < 64);
        bool vx1 = (x1i >= 0) && (x1i < 64);
        float w00 = (vy0 && vx0) ? wy0 * wx0 : 0.0f;
        float w01 = (vy0 && vx1) ? wy0 * wx1 : 0.0f;
        float w10 = (vy1 && vx0) ? wy1 * wx0 : 0.0f;
        float w11 = (vy1 && vx1) ? wy1 * wx1 : 0.0f;
        int yc0 = min(max(y0, 0), 63), yc1 = min(max(y1, 0), 63);
        int xc0 = min(max(x0i, 0), 63), xc1 = min(max(x1i, 0), 63);
        int i00 = yc0 * 64 + xc0, i01 = yc0 * 64 + xc1;
        int i10 = yc1 * 64 + xc0, i11 = yc1 * 64 + xc1;

        __syncthreads();   // previous iteration's fragment fully consumed
        #pragma unroll
        for (int j = 0; j < 4; ++j) {
            int e  = e0 + j;
            int K  = e + 16 * bhi;                 // B-matrix kmap
            int ch = g * 64 + c0 + K;
            const float* xb = x + (((size_t)b * 256 + ch) * 4096);
            float val = w00 * xb[i00] + w01 * xb[i01] + w10 * xb[i10] + w11 * xb[i11];
            sB[blane * 16 + e] = f2bf(val);
        }
        __syncthreads();

        v16bf a  = apack[((size_t)mt * KT_TOTAL + kt) * 32 + lane];
        v16bf bm = *(const v16bf*)(&sB[lane * 16]);
        acc = __builtin_amdgcn_wmma_f32_16x16x32_bf16(false, a, false, bm,
                                                      (short)0, acc, false, false);
    }

    int n = w0 + p;
    for (int v = 0; v < 8; ++v) {
        int oc = mt * 16 + 8 * hi + v;
        out[(((size_t)b * 256 + oc) * 64 + h) * 64 + n] = fmaxf(acc[v], 0.0f);
    }
}

// ---------------------------------------------------------------------------
extern "C" void kernel_launch(void* const* d_in, const int* in_sizes, int n_in,
                              void* d_out, int out_size, void* d_ws, size_t ws_size,
                              hipStream_t stream)
{
    const float* x  = (const float*)d_in[0];
    const float* w1 = (const float*)d_in[1];
    const float* b1 = (const float*)d_in[2];
    const float* w2 = (const float*)d_in[3];
    const float* b2 = (const float*)d_in[4];
    const float* wd = (const float*)d_in[5];
    float* out = (float*)d_out;

    // workspace layout
    char* ws = (char*)d_ws;
    float*  y1   = (float*)(ws);                                   // 33,554,432 B
    float*  offs = (float*)(ws + 33554432);                        //  9,437,184 B
    __bf16* pa1  = (__bf16*)(ws + 33554432 + 9437184);             //  1,179,648 B
    __bf16* pa2  = (__bf16*)(ws + 33554432 + 9437184 + 1179648);   //    589,824 B
    __bf16* pawd = (__bf16*)(ws + 33554432 + 9437184 + 1179648 + 589824); // 1,179,648 B

    // pack weights into WMMA A-fragment layout (bf16)
    int total1 = 16 * KT_TOTAL * 512;   // 256 outC
    int total2 = 8  * KT_TOTAL * 512;   // 72 outC padded to 128
    pack_wA<<<(total1 + 255) / 256, 256, 0, stream>>>(w1, pa1, 256, 16, 0);
    pack_wA<<<(total2 + 255) / 256, 256, 0, stream>>>(w2, pa2,  72,  8, 0);
    pack_wA<<<(total1 + 255) / 256, 256, 0, stream>>>(wd, pawd, 256, 16, 1);

    dim3 blk(128, 1, 1);
    // stage 1: y1 = conv(x, w1) + b1
    conv3x3_wmma<<<dim3(2048, 4, 1), blk, 0, stream>>>(x,  pa1, b1, y1, 256);
    // stage 2: offset = conv(y1, w2) + b2
    conv3x3_wmma<<<dim3(2048, 2, 1), blk, 0, stream>>>(y1, pa2, b2, offs, 72);
    // stage 3: out = relu(deform_conv(x, offset, wd))
    deform_wmma<<<dim3(2048, 4, 1), blk, 0, stream>>>(x, offs, pawd, out);
}
